// CaptionDecoder_4733053960287
// MI455X (gfx1250) — matrix operand order repaired and checked by reading, not compiled, not run on hardware
//
#include <hip/hip_runtime.h>
#include <hip/hip_bf16.h>

// ---------------------------------------------------------------- constants
static constexpr int B_   = 64;
static constexpr int N_   = 196;
static constexpr int ENC  = 2048;
static constexpr int DEC  = 512;
static constexpr int E_   = 512;
static constexpr int ATT  = 512;
static constexpr int V_   = 32000;
static constexpr int T_   = 32;
static constexpr int TS   = T_ - 1;          // 31 timesteps
static constexpr int KG   = E_ + ENC + DEC;  // 3072: fused gates K
static constexpr int NG   = 4 * DEC;         // 2048: gates N

// ---------------------------------------------------------------- types
typedef __attribute__((ext_vector_type(16))) __bf16 v16bf;
typedef __attribute__((ext_vector_type(8)))  __bf16 v8bf;
typedef __attribute__((ext_vector_type(8)))  float  v8f;

// bf16 <-> f32 via bit ops (round-to-nearest-even), no native __bf16 math needed
__device__ __forceinline__ __bf16 f2bf(float f) {
    union { float f; unsigned u; } x; x.f = f;
    unsigned r = (x.u + 0x7FFFu + ((x.u >> 16) & 1u)) >> 16;
    unsigned short s = (unsigned short)r;
    return __builtin_bit_cast(__bf16, s);
}
__device__ __forceinline__ float bf2f(__bf16 b) {
    unsigned short s = __builtin_bit_cast(unsigned short, b);
    union { unsigned u; float f; } x; x.u = ((unsigned)s) << 16;
    return x.f;
}
__device__ __forceinline__ float sigm(float x) { return 1.f / (1.f + __expf(-x)); }

// ---------------------------------------------------------------- generic WMMA GEMM
// C[M,N] = A[M,K](bf16 row-major) x Bt[N,K](bf16 row-major)^T + bias[N]
// 4 waves per block; each wave -> 16(M) x 32(N) tile via two accumulators.
// Requires: M%16==0, N%128==0, K%32==0.
__global__ void __launch_bounds__(128) gemm_bf16(
    const __bf16* __restrict__ A, const __bf16* __restrict__ Bt,
    const float* __restrict__ bias, float* __restrict__ C,
    int M, int N, int K, long long ldc)
{
    const int lane = threadIdx.x & 31;
    const int wv   = threadIdx.x >> 5;
    const int tm   = blockIdx.y * 16;
    const int tn   = blockIdx.x * 128 + wv * 32;
    const int l15  = lane & 15;
    const bool hi  = lane >= 16;

    const __bf16* arow  = A  + (size_t)(tm + l15) * K;
    const __bf16* brow0 = Bt + (size_t)(tn + l15) * K;
    const __bf16* brow1 = Bt + (size_t)(tn + 16 + l15) * K;
    const int aoff = hi ? 16 : 0;    // A: lanes 0-15 hold K [0..15], lanes 16-31 [16..31]
    const int boff = hi ? 16 : 0;    // B: lanes 0-15 hold K [0..15], lanes 16-31 [16..31]

    v8f acc0 = {}; v8f acc1 = {};
    for (int k = 0; k < K; k += 32) {
        union { v16bf v; v8bf h[2]; } a;
        a.h[0] = *(const v8bf*)(arow + k + aoff);
        a.h[1] = *(const v8bf*)(arow + k + 8 + aoff);
        v16bf b0 = *(const v16bf*)(brow0 + k + boff);
        v16bf b1 = *(const v16bf*)(brow1 + k + boff);
        acc0 = __builtin_amdgcn_wmma_f32_16x16x32_bf16(false, a.v, false, b0,
                                                       (short)0, acc0, false, false);
        acc1 = __builtin_amdgcn_wmma_f32_16x16x32_bf16(false, a.v, false, b1,
                                                       (short)0, acc1, false, false);
    }

    // C/D layout: VGPR r, lanes 0-15 -> M=tm+r, lanes 16-31 -> M=tm+8+r; N = tile_n + (lane&15)
    const int n0 = tn + l15;
    const int n1 = tn + 16 + l15;
    const int mb = tm + (hi ? 8 : 0);
    const float bv0 = bias ? bias[n0] : 0.f;
    const float bv1 = bias ? bias[n1] : 0.f;
#pragma unroll
    for (int r = 0; r < 8; ++r) {
        C[(size_t)(mb + r) * ldc + n0] = acc0[r] + bv0;
        C[(size_t)(mb + r) * ldc + n1] = acc1[r] + bv1;
    }
}

// ---------------------------------------------------------------- setup kernels
__global__ void f32_to_bf16_k(const float* __restrict__ in, __bf16* __restrict__ out, size_t n) {
    size_t i = (size_t)blockIdx.x * blockDim.x + threadIdx.x;
    size_t st = (size_t)gridDim.x * blockDim.x;
    for (; i < n; i += st) out[i] = f2bf(in[i]);
}

// fp32 [R,C] -> bf16 [C,R] (tiled, LDS, coalesced both sides)
__global__ void transpose_to_bf16_k(const float* __restrict__ in, __bf16* __restrict__ out,
                                    int R, int C) {
    __shared__ float tile[32][33];
    int c0 = blockIdx.x * 32, r0 = blockIdx.y * 32;
    for (int i = threadIdx.y; i < 32; i += 8) {
        int r = r0 + i, c = c0 + threadIdx.x;
        tile[i][threadIdx.x] = (r < R && c < C) ? in[(size_t)r * C + c] : 0.f;
    }
    __syncthreads();
    for (int i = threadIdx.y; i < 32; i += 8) {
        int r = r0 + threadIdx.x, c = c0 + i;
        if (r < R && c < C) out[(size_t)c * R + r] = f2bf(tile[threadIdx.x][i]);
    }
}

// fused gate weight: Wg[n, 0:2560]=W_ih[n,:], Wg[n, 2560:3072]=W_hh[n,:]  (both already [N,K])
__global__ void build_wg_k(const float* __restrict__ W_ih, const float* __restrict__ W_hh,
                           __bf16* __restrict__ Wg) {
    size_t i = (size_t)blockIdx.x * blockDim.x + threadIdx.x;
    size_t st = (size_t)gridDim.x * blockDim.x;
    size_t tot = (size_t)NG * KG;
    for (; i < tot; i += st) {
        int n = (int)(i / KG), k = (int)(i % KG);
        float v = (k < E_ + ENC) ? W_ih[(size_t)n * (E_ + ENC) + k]
                                 : W_hh[(size_t)n * DEC + (k - (E_ + ENC))];
        Wg[i] = f2bf(v);
    }
}

__global__ void bias_sum_k(const float* __restrict__ a, const float* __restrict__ b,
                           float* __restrict__ o, int n) {
    int i = blockIdx.x * blockDim.x + threadIdx.x;
    if (i < n) o[i] = a[i] + b[i];
}

// gather all caption embeddings to bf16: xs_bf[t][b][e] = emb[captions[b,t], e], t<31
__global__ void gather_xs_k(const int* __restrict__ captions, const float* __restrict__ emb,
                            __bf16* __restrict__ xs_bf) {
    size_t i = (size_t)blockIdx.x * blockDim.x + threadIdx.x;
    size_t st = (size_t)gridDim.x * blockDim.x;
    size_t tot = (size_t)TS * B_ * E_;
    for (; i < tot; i += st) {
        int t = (int)(i / (B_ * E_));
        int rem = (int)(i % (B_ * E_));
        int b = rem / E_, e = rem % E_;
        int tok = captions[b * T_ + t];
        xs_bf[i] = f2bf(emb[(size_t)tok * E_ + e]);
    }
}

__global__ void __launch_bounds__(256) mean_pool_k(const float* __restrict__ features,
                                                   __bf16* __restrict__ mean_bf) {
    int b = blockIdx.x;
    for (int d = threadIdx.x; d < ENC; d += 256) {
        const float* f = features + (size_t)b * N_ * ENC + d;
        float s = 0.f;
        for (int n = 0; n < N_; ++n) s += f[(size_t)n * ENC];
        mean_bf[b * ENC + d] = f2bf(s * (1.f / (float)N_));
    }
}

// h0 fp32 -> h_bf and Abuf[:, 2560:3072]
__global__ void hc_init_k(const float* __restrict__ h0, __bf16* __restrict__ h_bf,
                          __bf16* __restrict__ Abuf) {
    int idx = blockIdx.x * 256 + threadIdx.x;
    if (idx >= B_ * DEC) return;
    int b = idx >> 9, d = idx & 511;
    __bf16 v = f2bf(h0[idx]);
    h_bf[idx] = v;
    Abuf[b * KG + (E_ + ENC) + d] = v;
}

// ---------------------------------------------------------------- per-step kernels
// one block per batch row: e = tanh(enc_att + hdec)·v, softmax, context; fills Abuf x+context
__global__ void __launch_bounds__(256) attn_step_k(
    const float* __restrict__ enc_att, const float* __restrict__ hdec,
    const float* __restrict__ att_v, const __bf16* __restrict__ feat_bf,
    const __bf16* __restrict__ xs_t, __bf16* __restrict__ Abuf,
    float* __restrict__ alphas, int t)
{
    const int b = blockIdx.x;
    const int tid = threadIdx.x;
    __shared__ float e_sh[N_];
    __shared__ float red[256];

    // copy this step's embedding into Abuf[:, 0:512]
    for (int d = tid; d < E_; d += 256)
        Abuf[b * KG + d] = xs_t[b * E_ + d];

    // e[n] = sum_d tanh(enc_att[b,n,d] + hdec[b,d]) * v[d]   (one wave per row group)
    const int lane = tid & 31, wid = tid >> 5;
    for (int n = wid; n < N_; n += 8) {
        const float* ea = enc_att + ((size_t)b * N_ + n) * ATT;
        float s = 0.f;
        for (int d = lane; d < ATT; d += 32)
            s += tanhf(ea[d] + hdec[b * ATT + d]) * att_v[d];
        for (int off = 16; off; off >>= 1) s += __shfl_down(s, off, 32);
        if (lane == 0) e_sh[n] = s;
    }
    __syncthreads();

    // softmax over 196 spatial locations
    float v = (tid < N_) ? e_sh[tid] : -3.4e38f;
    red[tid] = v; __syncthreads();
    for (int s = 128; s; s >>= 1) { if (tid < s) red[tid] = fmaxf(red[tid], red[tid + s]); __syncthreads(); }
    float mx = red[0]; __syncthreads();
    float ex = (tid < N_) ? __expf(v - mx) : 0.f;
    red[tid] = ex; __syncthreads();
    for (int s = 128; s; s >>= 1) { if (tid < s) red[tid] += red[tid + s]; __syncthreads(); }
    float inv = 1.f / red[0]; __syncthreads();
    if (tid < N_) {
        float a = ex * inv;
        e_sh[tid] = a;
        alphas[((size_t)b * TS + t) * N_ + tid] = a;
    }
    __syncthreads();

    // context[b,d] = sum_n alpha[n] * features[b,n,d]  -> Abuf[:, 512:2560] (bf16)
    for (int d = tid; d < ENC; d += 256) {
        const __bf16* fb = feat_bf + (size_t)b * N_ * ENC + d;
        float acc = 0.f;
        for (int n = 0; n < N_; ++n) acc += e_sh[n] * bf2f(fb[(size_t)n * ENC]);
        Abuf[b * KG + E_ + d] = f2bf(acc);
    }
}

// LSTM pointwise: torch gate order i,f,g,o; updates c, writes bf16 h (+ into Abuf)
__global__ void __launch_bounds__(256) lstm_pointwise_k(
    const float* __restrict__ gates, float* __restrict__ c,
    __bf16* __restrict__ h_bf, __bf16* __restrict__ Abuf)
{
    int idx = blockIdx.x * 256 + threadIdx.x;
    if (idx >= B_ * DEC) return;
    int b = idx >> 9, d = idx & 511;
    const float* g = gates + (size_t)b * NG;
    float ig = g[d], fg = g[DEC + d], gg = g[2 * DEC + d], og = g[3 * DEC + d];
    float cn = sigm(fg) * c[idx] + sigm(ig) * tanhf(gg);
    c[idx] = cn;
    float h = sigm(og) * tanhf(cn);
    __bf16 hb = f2bf(h);
    h_bf[idx] = hb;
    Abuf[b * KG + (E_ + ENC) + d] = hb;
}

// ---------------------------------------------------------------- host
extern "C" void kernel_launch(void* const* d_in, const int* in_sizes, int n_in,
                              void* d_out, int out_size, void* d_ws, size_t ws_size,
                              hipStream_t stream) {
    const float* features = (const float*)d_in[0];
    const int*   captions = (const int*)  d_in[1];
    const float* emb      = (const float*)d_in[2];
    const float* att_Wenc = (const float*)d_in[3];
    const float* att_Wdec = (const float*)d_in[4];
    const float* att_v    = (const float*)d_in[5];
    const float* init_h_W = (const float*)d_in[6];
    const float* init_h_b = (const float*)d_in[7];
    const float* init_c_W = (const float*)d_in[8];
    const float* init_c_b = (const float*)d_in[9];
    const float* W_ih     = (const float*)d_in[10];
    const float* W_hh     = (const float*)d_in[11];
    const float* b_ih     = (const float*)d_in[12];
    const float* b_hh     = (const float*)d_in[13];
    const float* out_W    = (const float*)d_in[14];
    const float* out_b    = (const float*)d_in[15];

    float* preds  = (float*)d_out;                         // [B, 31, V]
    float* alphas = preds + (size_t)B_ * TS * V_;          // [B, 31, N]

    // ---- workspace bump allocator
    char* p = (char*)d_ws;
    auto alloc = [&](size_t bytes) -> void* {
        void* r = (void*)p; p += (bytes + 255) & ~(size_t)255; return r;
    };
    __bf16* feat_bf  = (__bf16*)alloc((size_t)B_ * N_ * ENC * 2);   // 51.4 MB
    float*  enc_att  = (float*) alloc((size_t)B_ * N_ * ATT * 4);   // 25.7 MB
    __bf16* outWT    = (__bf16*)alloc((size_t)V_ * DEC * 2);        // 32.8 MB
    __bf16* Wg       = (__bf16*)alloc((size_t)NG * KG * 2);         // 12.6 MB
    __bf16* attWencT = (__bf16*)alloc((size_t)ATT * ENC * 2);
    __bf16* attWdecT = (__bf16*)alloc((size_t)ATT * DEC * 2);
    __bf16* initHWT  = (__bf16*)alloc((size_t)DEC * ENC * 2);
    __bf16* initCWT  = (__bf16*)alloc((size_t)DEC * ENC * 2);
    __bf16* xs_bf    = (__bf16*)alloc((size_t)TS * B_ * E_ * 2);
    __bf16* mean_bf  = (__bf16*)alloc((size_t)B_ * ENC * 2);
    float*  h0tmp    = (float*) alloc((size_t)B_ * DEC * 4);
    float*  cbuf     = (float*) alloc((size_t)B_ * DEC * 4);
    float*  hdec     = (float*) alloc((size_t)B_ * ATT * 4);
    __bf16* h_bf     = (__bf16*)alloc((size_t)B_ * DEC * 2);
    __bf16* Abuf     = (__bf16*)alloc((size_t)B_ * KG * 2);
    float*  gates    = (float*) alloc((size_t)B_ * NG * 4);
    float*  bsum     = (float*) alloc((size_t)NG * 4);
    (void)ws_size; (void)n_in; (void)in_sizes; (void)out_size;

    auto gemm = [&](const __bf16* A, const __bf16* Bt, const float* bias, float* C,
                    int M, int N, int K, long long ldc) {
        dim3 g(N / 128, M / 16);
        gemm_bf16<<<g, 128, 0, stream>>>(A, Bt, bias, C, M, N, K, ldc);
    };

    // ---- setup (runs every call; deterministic)
    f32_to_bf16_k<<<4096, 256, 0, stream>>>(features, feat_bf, (size_t)B_ * N_ * ENC);
    {
        dim3 blk(32, 8);
        transpose_to_bf16_k<<<dim3(ATT / 32, ENC / 32), blk, 0, stream>>>(att_Wenc, attWencT, ENC, ATT);
        transpose_to_bf16_k<<<dim3(ATT / 32, DEC / 32), blk, 0, stream>>>(att_Wdec, attWdecT, DEC, ATT);
        transpose_to_bf16_k<<<dim3(DEC / 32, ENC / 32), blk, 0, stream>>>(init_h_W, initHWT, ENC, DEC);
        transpose_to_bf16_k<<<dim3(DEC / 32, ENC / 32), blk, 0, stream>>>(init_c_W, initCWT, ENC, DEC);
        transpose_to_bf16_k<<<dim3(V_ / 32, DEC / 32),  blk, 0, stream>>>(out_W,    outWT,   DEC, V_);
    }
    build_wg_k<<<4096, 256, 0, stream>>>(W_ih, W_hh, Wg);
    bias_sum_k<<<(NG + 255) / 256, 256, 0, stream>>>(b_ih, b_hh, bsum, NG);
    gather_xs_k<<<2048, 256, 0, stream>>>(captions, emb, xs_bf);
    mean_pool_k<<<B_, 256, 0, stream>>>(features, mean_bf);

    // h0 = mean_f @ init_h_W + b ; c0 = mean_f @ init_c_W + b
    gemm(mean_bf, initHWT, init_h_b, h0tmp, B_, DEC, ENC, DEC);
    gemm(mean_bf, initCWT, init_c_b, cbuf,  B_, DEC, ENC, DEC);
    hc_init_k<<<(B_ * DEC + 255) / 256, 256, 0, stream>>>(h0tmp, h_bf, Abuf);

    // enc_att = features @ att_Wenc  (loop-invariant)
    gemm(feat_bf, attWencT, nullptr, enc_att, B_ * N_, ATT, ENC, ATT);

    // ---- recurrence: 31 sequential steps
    for (int t = 0; t < TS; ++t) {
        // hdec = h @ att_Wdec
        gemm(h_bf, attWdecT, nullptr, hdec, B_, ATT, DEC, ATT);
        // attention + softmax + context (fills Abuf[:,0:2560], writes alphas)
        attn_step_k<<<B_, 256, 0, stream>>>(enc_att, hdec, att_v, feat_bf,
                                            xs_bf + (size_t)t * B_ * E_, Abuf, alphas, t);
        // gates = [x|context|h] @ [W_ih|W_hh]^T + (b_ih + b_hh)
        gemm(Abuf, Wg, bsum, gates, B_, NG, KG, NG);
        // LSTM pointwise update (writes new bf16 h into h_bf and Abuf)
        lstm_pointwise_k<<<(B_ * DEC + 255) / 256, 256, 0, stream>>>(gates, cbuf, h_bf, Abuf);
        // preds[:, t, :] = h @ out_W + out_b   (strided store into [B,31,V])
        gemm(h_bf, outWT, out_b, preds + (size_t)t * V_, B_, V_, DEC, (long long)TS * V_);
    }
}
